// TransformerDecLayer_51591147160167
// MI455X (gfx1250) — matrix operand order repaired and checked by reading, not compiled, not run on hardware
//
#include <hip/hip_runtime.h>

// ---------------------------------------------------------------------------
// Problem constants (from reference): B=8, SQ=512, SK=1024, D=1024, H=16,
// F=4096, DK=64.
// ---------------------------------------------------------------------------
#define Bc  8
#define Hh  16
#define SQc 512
#define SKc 1024
#define Dc  1024
#define Fc  4096
#define DKc 64

typedef _Float16 half16v __attribute__((ext_vector_type(16)));
typedef _Float16 half8v  __attribute__((ext_vector_type(8)));
typedef float    f32x4   __attribute__((ext_vector_type(4)));
typedef float    v8f     __attribute__((ext_vector_type(8)));

#define DEV __device__ __forceinline__

// ---------------------------------------------------------------------------
// WMMA wrapper: D = A(16x32 f16) * B(32x16 f16) + C(16x16 f32)   [wave32]
// ---------------------------------------------------------------------------
DEV v8f wmma16(half16v a, half16v b, v8f c) {
  return __builtin_amdgcn_wmma_f32_16x16x32_f16(
      /*neg_a=*/false, a, /*neg_b=*/false, b,
      /*c_mod=*/(short)0, c, /*reuse_a=*/false, /*reuse_b=*/false);
}

// A-matrix fragment (16x32 f16) loaded from an fp32 row-major matrix.
// Layout: lane<16 holds row M=lane with K = 0..15;
//         lane>=16 holds row M=lane-16 with K = 16..31 (contiguous, matches B).
DEV half16v load_a_f32(const float* __restrict__ base, int ld, int k0, int lane) {
  const float* p = base + (size_t)(lane & 15) * ld + k0 + ((lane >> 4) << 4);
  f32x4 f0 = *(const f32x4*)(p + 0);
  f32x4 f1 = *(const f32x4*)(p + 4);
  f32x4 f2 = *(const f32x4*)(p + 8);
  f32x4 f3 = *(const f32x4*)(p + 12);
  half16v a;
  a[0]  = (_Float16)f0[0]; a[1]  = (_Float16)f0[1]; a[2]  = (_Float16)f0[2]; a[3]  = (_Float16)f0[3];
  a[4]  = (_Float16)f1[0]; a[5]  = (_Float16)f1[1]; a[6]  = (_Float16)f1[2]; a[7]  = (_Float16)f1[3];
  a[8]  = (_Float16)f2[0]; a[9]  = (_Float16)f2[1]; a[10] = (_Float16)f2[2]; a[11] = (_Float16)f2[3];
  a[12] = (_Float16)f3[0]; a[13] = (_Float16)f3[1]; a[14] = (_Float16)f3[2]; a[15] = (_Float16)f3[3];
  return a;
}

// B-matrix fragment (32x16 f16). "base" is a row-major [N,K] matrix whose
// rows are the columns of B. Layout used here (same k order as A): lane<16 holds col N=lane, K=0..15;
// lane>=16 holds col N=lane-16, K=16..31 (contiguous 16 halfs).
DEV half16v load_b_f16(const _Float16* __restrict__ base, int ld, int k0, int lane) {
  const _Float16* p = base + (size_t)(lane & 15) * ld + k0 + ((lane >> 4) << 4);
  half8v lo = *(const half8v*)(p + 0);
  half8v hi = *(const half8v*)(p + 8);
  return __builtin_shufflevector(lo, hi, 0, 1, 2, 3, 4, 5, 6, 7,
                                 8, 9, 10, 11, 12, 13, 14, 15);
}

// Same as load_b_f16 but source rows are fp32 (converted in-register).
DEV half16v load_b_f32(const float* __restrict__ base, int ld, int k0, int lane) {
  const float* p = base + (size_t)(lane & 15) * ld + k0 + ((lane >> 4) << 4);
  f32x4 f0 = *(const f32x4*)(p + 0);
  f32x4 f1 = *(const f32x4*)(p + 4);
  f32x4 f2 = *(const f32x4*)(p + 8);
  f32x4 f3 = *(const f32x4*)(p + 12);
  half16v b;
  b[0]  = (_Float16)f0[0]; b[1]  = (_Float16)f0[1]; b[2]  = (_Float16)f0[2]; b[3]  = (_Float16)f0[3];
  b[4]  = (_Float16)f1[0]; b[5]  = (_Float16)f1[1]; b[6]  = (_Float16)f1[2]; b[7]  = (_Float16)f1[3];
  b[8]  = (_Float16)f2[0]; b[9]  = (_Float16)f2[1]; b[10] = (_Float16)f2[2]; b[11] = (_Float16)f2[3];
  b[12] = (_Float16)f3[0]; b[13] = (_Float16)f3[1]; b[14] = (_Float16)f3[2]; b[15] = (_Float16)f3[3];
  return b;
}

// C/D layout: lanes 0-15 -> N=lane, M=vgpr r; lanes 16-31 -> M=r+8.
DEV void gemm_store(float* __restrict__ C, int N, int m0, int n0, int lane,
                    v8f acc, const float* __restrict__ bias, int relu) {
  int n = n0 + (lane & 15);
  float bv = bias[n];
  int mb = m0 + ((lane >> 4) << 3);
#pragma unroll
  for (int r = 0; r < 8; ++r) {
    float v = acc[r] + bv;
    if (relu) v = fmaxf(v, 0.f);
    C[(size_t)(mb + r) * N + n] = v;
  }
}

// ---------------------------------------------------------------------------
// GEMM: C[M,N] = act(A[M,K](fp32) @ Bt[N,K](f16, pre-transposed W) + bias)
// Block = 256 threads = 8 waves; wave tile 32x64 (2x4 WMMA, A-frags reused
// 4x to amortize the fp32->f16 conversion); block tile 64x256.
// ---------------------------------------------------------------------------
__global__ __launch_bounds__(256) void gemm_wmma_kernel(
    const float* __restrict__ A, const _Float16* __restrict__ Bt,
    const float* __restrict__ bias, float* __restrict__ C,
    int M, int N, int K, int relu) {
  int lane = threadIdx.x & 31;
  int wave = threadIdx.x >> 5;
  int m0 = blockIdx.x * 64 + (wave & 1) * 32;
  int n0 = blockIdx.y * 256 + (wave >> 1) * 64;
  const float*    A0 = A + (size_t)m0 * K;
  const float*    A1 = A + (size_t)(m0 + 16) * K;
  const _Float16* B0 = Bt + (size_t)n0 * K;
  const _Float16* B1 = Bt + (size_t)(n0 + 16) * K;
  const _Float16* B2 = Bt + (size_t)(n0 + 32) * K;
  const _Float16* B3 = Bt + (size_t)(n0 + 48) * K;

  v8f c00 = {}, c01 = {}, c02 = {}, c03 = {};
  v8f c10 = {}, c11 = {}, c12 = {}, c13 = {};
  for (int k0 = 0; k0 < K; k0 += 32) {
    if (k0 + 64 < K) {  // global_prefetch_b8 of the next-next k-tile
      __builtin_prefetch(A0 + (size_t)(lane & 15) * K + k0 + 64, 0, 1);
      __builtin_prefetch(B0 + (size_t)(lane & 15) * K + k0 + 64, 0, 1);
    }
    half16v a0 = load_a_f32(A0, K, k0, lane);
    half16v a1 = load_a_f32(A1, K, k0, lane);
    half16v b0 = load_b_f16(B0, K, k0, lane);
    half16v b1 = load_b_f16(B1, K, k0, lane);
    half16v b2 = load_b_f16(B2, K, k0, lane);
    half16v b3 = load_b_f16(B3, K, k0, lane);
    c00 = wmma16(a0, b0, c00);
    c01 = wmma16(a0, b1, c01);
    c02 = wmma16(a0, b2, c02);
    c03 = wmma16(a0, b3, c03);
    c10 = wmma16(a1, b0, c10);
    c11 = wmma16(a1, b1, c11);
    c12 = wmma16(a1, b2, c12);
    c13 = wmma16(a1, b3, c13);
  }
  gemm_store(C, N, m0,      n0,      lane, c00, bias, relu);
  gemm_store(C, N, m0,      n0 + 16, lane, c01, bias, relu);
  gemm_store(C, N, m0,      n0 + 32, lane, c02, bias, relu);
  gemm_store(C, N, m0,      n0 + 48, lane, c03, bias, relu);
  gemm_store(C, N, m0 + 16, n0,      lane, c10, bias, relu);
  gemm_store(C, N, m0 + 16, n0 + 16, lane, c11, bias, relu);
  gemm_store(C, N, m0 + 16, n0 + 32, lane, c12, bias, relu);
  gemm_store(C, N, m0 + 16, n0 + 48, lane, c13, bias, relu);
}

// ---------------------------------------------------------------------------
// Attention scores: S[z, q, k] = (Q_z[q,:] . K_z[k,:]) / 8, with causal or
// encoder-mask applied. Q/K are the flat [B*S, D] projection outputs; head
// slices are contiguous 64 floats so fragments load directly (no reshape).
// Block tile 64(q) x 128(k); grid.z = b*H + h. (Only 2 k-iterations; this
// kernel is store-bound, so the 2x2 wave tile is kept.)
// ---------------------------------------------------------------------------
__global__ __launch_bounds__(256) void attn_scores_kernel(
    const float* __restrict__ Q, const float* __restrict__ Kt,
    const int* __restrict__ mask, float* __restrict__ S,
    int SQ, int SKt, int causal) {
  int lane = threadIdx.x & 31;
  int wave = threadIdx.x >> 5;
  int z = blockIdx.z;
  int b = z >> 4;   // H == 16
  int h = z & 15;
  int q0 = blockIdx.x * 64 + (wave & 1) * 32;
  int n0 = blockIdx.y * 128 + (wave >> 1) * 32;
  const float* Qb = Q + ((size_t)b * SQ) * Dc + h * DKc;
  const float* Kb = Kt + ((size_t)b * SKt) * Dc + h * DKc;

  v8f acc00 = {}, acc01 = {}, acc10 = {}, acc11 = {};
#pragma unroll
  for (int kk = 0; kk < DKc; kk += 32) {
    half16v a0 = load_a_f32(Qb + (size_t)q0 * Dc,        Dc, kk, lane);
    half16v a1 = load_a_f32(Qb + (size_t)(q0 + 16) * Dc, Dc, kk, lane);
    half16v b0 = load_b_f32(Kb + (size_t)n0 * Dc,        Dc, kk, lane);
    half16v b1 = load_b_f32(Kb + (size_t)(n0 + 16) * Dc, Dc, kk, lane);
    acc00 = wmma16(a0, b0, acc00);
    acc01 = wmma16(a0, b1, acc01);
    acc10 = wmma16(a1, b0, acc10);
    acc11 = wmma16(a1, b1, acc11);
  }

  float* Sb = S + (size_t)z * SQ * SKt;
  const int* maskb = mask ? (mask + (size_t)b * SKt) : nullptr;
  const v8f* accs[4] = {&acc00, &acc01, &acc10, &acc11};
#pragma unroll
  for (int t = 0; t < 4; ++t) {
    int ti = t >> 1, tj = t & 1;
    int n = n0 + tj * 16 + (lane & 15);
    int mb = q0 + ti * 16 + ((lane >> 4) << 3);
    bool masked = (maskb != nullptr) && (maskb[n] == 0);
#pragma unroll
    for (int r = 0; r < 8; ++r) {
      int m = mb + r;
      float v = (*accs[t])[r] * 0.125f;  // 1/sqrt(64)
      if ((causal && n > m) || masked) v = -__builtin_inff();
      Sb[(size_t)m * SKt + n] = v;
    }
  }
}

// ---------------------------------------------------------------------------
// AV: Ctx[b, q, h*64+dk] = sum_k P[z,q,k] * V[z,k,dk].
// P fp32 (converted in-register), Vt f16 [z, 64, S] (pre-transposed).
// Wave tile 32(q) x 64(dk) = full head width; block tile 256x64.
// ---------------------------------------------------------------------------
__global__ __launch_bounds__(256) void attn_av_kernel(
    const float* __restrict__ P, const _Float16* __restrict__ Vt,
    float* __restrict__ Ctx, int SQ, int SKt) {
  int lane = threadIdx.x & 31;
  int wave = threadIdx.x >> 5;
  int z = blockIdx.z;
  int b = z >> 4;
  int h = z & 15;
  int m0 = blockIdx.x * 256 + wave * 32;
  const float*    Pb = P + (size_t)z * SQ * SKt;
  const _Float16* V0 = Vt + (size_t)z * DKc * SKt;
  const _Float16* V1 = V0 + (size_t)16 * SKt;
  const _Float16* V2 = V0 + (size_t)32 * SKt;
  const _Float16* V3 = V0 + (size_t)48 * SKt;

  v8f c00 = {}, c01 = {}, c02 = {}, c03 = {};
  v8f c10 = {}, c11 = {}, c12 = {}, c13 = {};
  for (int k0 = 0; k0 < SKt; k0 += 32) {
    half16v a0 = load_a_f32(Pb + (size_t)m0 * SKt,        SKt, k0, lane);
    half16v a1 = load_a_f32(Pb + (size_t)(m0 + 16) * SKt, SKt, k0, lane);
    half16v b0 = load_b_f16(V0, SKt, k0, lane);
    half16v b1 = load_b_f16(V1, SKt, k0, lane);
    half16v b2 = load_b_f16(V2, SKt, k0, lane);
    half16v b3 = load_b_f16(V3, SKt, k0, lane);
    c00 = wmma16(a0, b0, c00);
    c01 = wmma16(a0, b1, c01);
    c02 = wmma16(a0, b2, c02);
    c03 = wmma16(a0, b3, c03);
    c10 = wmma16(a1, b0, c10);
    c11 = wmma16(a1, b1, c11);
    c12 = wmma16(a1, b2, c12);
    c13 = wmma16(a1, b3, c13);
  }
  const v8f* accs[8] = {&c00, &c01, &c02, &c03, &c10, &c11, &c12, &c13};
#pragma unroll
  for (int t = 0; t < 8; ++t) {
    int ti = t >> 2, tj = t & 3;
    int n = tj * 16 + (lane & 15);
    int mb = m0 + ti * 16 + ((lane >> 4) << 3);
#pragma unroll
    for (int r = 0; r < 8; ++r)
      Ctx[((size_t)b * SQ + mb + r) * Dc + h * DKc + n] = (*accs[t])[r];
  }
}

// ---------------------------------------------------------------------------
// Transpose + fp32->f16 convert: W[K,N] (row-major fp32) -> Wt[N,K] f16.
// LDS 32x33 tile keeps both global streams coalesced.
// ---------------------------------------------------------------------------
__global__ __launch_bounds__(256) void transpose_cvt_kernel(
    const float* __restrict__ in, _Float16* __restrict__ out, int K, int N) {
  __shared__ float tile[32][33];
  int tx = threadIdx.x & 31;
  int ty = threadIdx.x >> 5;
  int kt = blockIdx.x * 32;
  int nt = blockIdx.y * 32;
#pragma unroll
  for (int r = ty; r < 32; r += 8)
    tile[r][tx] = in[(size_t)(kt + r) * N + nt + tx];
  __syncthreads();
#pragma unroll
  for (int r = ty; r < 32; r += 8)
    out[(size_t)(nt + r) * K + kt + tx] = (_Float16)tile[tx][r];
}

// V[B*S, D] fp32 -> Vt[z, dk, s] f16 (per-head transpose for AV B-fragments).
__global__ __launch_bounds__(256) void vtranspose_kernel(
    const float* __restrict__ V, _Float16* __restrict__ Vt, int S) {
  __shared__ float tile[32][33];
  int tx = threadIdx.x & 31;
  int ty = threadIdx.x >> 5;
  int z = blockIdx.z;
  int b = z >> 4;
  int h = z & 15;
  int st = blockIdx.x * 32;
  int dt = blockIdx.y * 32;
#pragma unroll
  for (int r = ty; r < 32; r += 8)
    tile[r][tx] = V[((size_t)b * S + st + r) * Dc + h * DKc + dt + tx];
  __syncthreads();
#pragma unroll
  for (int r = ty; r < 32; r += 8)
    Vt[((size_t)z * DKc + dt + r) * S + st + tx] = (_Float16)tile[tx][r];
}

// ---------------------------------------------------------------------------
// Row softmax, in place. One wave32 per row, row length LEN (512 or 1024).
// ---------------------------------------------------------------------------
template <int LEN>
__global__ __launch_bounds__(256) void softmax_kernel(float* __restrict__ S) {
  int lane = threadIdx.x & 31;
  int wave = threadIdx.x >> 5;
  size_t row = (size_t)blockIdx.x * 8 + wave;
  float* p = S + row * LEN;
  constexpr int C = LEN / 32;
  float v[C];
  float mx = -__builtin_inff();
#pragma unroll
  for (int i = 0; i < C; ++i) {
    v[i] = p[lane + i * 32];
    mx = fmaxf(mx, v[i]);
  }
#pragma unroll
  for (int off = 16; off > 0; off >>= 1)
    mx = fmaxf(mx, __shfl_xor(mx, off, 32));
  float s = 0.f;
#pragma unroll
  for (int i = 0; i < C; ++i) {
    v[i] = __expf(v[i] - mx);
    s += v[i];
  }
#pragma unroll
  for (int off = 16; off > 0; off >>= 1) s += __shfl_xor(s, off, 32);
  float inv = 1.f / s;
#pragma unroll
  for (int i = 0; i < C; ++i) p[lane + i * 32] = v[i] * inv;
}

// ---------------------------------------------------------------------------
// O = LayerNorm(X + R) * g + beta, row length 1024. One wave32 per row.
// ---------------------------------------------------------------------------
__global__ __launch_bounds__(256) void add_ln_kernel(
    const float* __restrict__ X, const float* __restrict__ R,
    const float* __restrict__ g, const float* __restrict__ be,
    float* __restrict__ O) {
  int lane = threadIdx.x & 31;
  int wave = threadIdx.x >> 5;
  size_t row = (size_t)blockIdx.x * 8 + wave;
  const float* xp = X + row * Dc;
  const float* rp = R + row * Dc;
  float v[32];
  float s = 0.f;
#pragma unroll
  for (int i = 0; i < 32; ++i) {
    int c = lane + i * 32;
    v[i] = xp[c] + rp[c];
    s += v[i];
  }
#pragma unroll
  for (int off = 16; off > 0; off >>= 1) s += __shfl_xor(s, off, 32);
  float mu = s * (1.f / 1024.f);
  float vs = 0.f;
#pragma unroll
  for (int i = 0; i < 32; ++i) {
    float d = v[i] - mu;
    vs += d * d;
  }
#pragma unroll
  for (int off = 16; off > 0; off >>= 1) vs += __shfl_xor(vs, off, 32);
  float rstd = rsqrtf(vs * (1.f / 1024.f) + 1e-5f);
#pragma unroll
  for (int i = 0; i < 32; ++i) {
    int c = lane + i * 32;
    O[row * Dc + c] = (v[i] - mu) * rstd * g[c] + be[c];
  }
}

// ---------------------------------------------------------------------------
// Host-side orchestration.
// ---------------------------------------------------------------------------
extern "C" void kernel_launch(void* const* d_in, const int* in_sizes, int n_in,
                              void* d_out, int out_size, void* d_ws,
                              size_t ws_size, hipStream_t stream) {
  (void)in_sizes; (void)n_in; (void)out_size; (void)ws_size;
  const float* x     = (const float*)d_in[0];
  const float* enc_o = (const float*)d_in[1];
  const int*   encm  = (const int*)d_in[2];
  const float* a1_wq = (const float*)d_in[3],  *a1_bq = (const float*)d_in[4];
  const float* a1_wk = (const float*)d_in[5],  *a1_bk = (const float*)d_in[6];
  const float* a1_wv = (const float*)d_in[7],  *a1_bv = (const float*)d_in[8];
  const float* a1_wo = (const float*)d_in[9],  *a1_bo = (const float*)d_in[10];
  const float* a2_wq = (const float*)d_in[11], *a2_bq = (const float*)d_in[12];
  const float* a2_wk = (const float*)d_in[13], *a2_bk = (const float*)d_in[14];
  const float* a2_wv = (const float*)d_in[15], *a2_bv = (const float*)d_in[16];
  const float* a2_wo = (const float*)d_in[17], *a2_bo = (const float*)d_in[18];
  const float* ff_w1 = (const float*)d_in[19], *ff_b1 = (const float*)d_in[20];
  const float* ff_w2 = (const float*)d_in[21], *ff_b2 = (const float*)d_in[22];
  const float* ln1_g = (const float*)d_in[23], *ln1_b = (const float*)d_in[24];
  const float* ln2_g = (const float*)d_in[25], *ln2_b = (const float*)d_in[26];
  const float* ln3_g = (const float*)d_in[27], *ln3_b = (const float*)d_in[28];

  float* out_x    = (float*)d_out;                         // [B,SQ,D]
  float* out_attn = out_x + (size_t)Bc * SQc * Dc;         // [B,H,SQ,SK]

  // Workspace layout (256B-aligned regions).
  char* ws = (char*)d_ws;
  size_t off = 0;
  auto take = [&](size_t bytes) -> char* {
    char* p = ws + off;
    off += (bytes + 255) & ~(size_t)255;
    return p;
  };
  _Float16* wt  = (_Float16*)take((size_t)Fc * Dc * 2);          // 8 MB  (max Wt)
  float*    Qb  = (float*)take((size_t)Bc * SQc * Dc * 4);       // 16 MB
  float*    Kb  = (float*)take((size_t)Bc * SKc * Dc * 4);       // 32 MB
  float*    Vb  = (float*)take((size_t)Bc * SKc * Dc * 4);       // 32 MB
  _Float16* vt  = (_Float16*)take((size_t)Bc * Hh * DKc * SKc * 2); // 16 MB
  float*    Sb  = (float*)take((size_t)Bc * Hh * SQc * SQc * 4); // 128 MB (self scores)
  float*    ctx = (float*)take((size_t)Bc * SQc * Dc * 4);       // 16 MB
  float*    res = (float*)take((size_t)Bc * SQc * Dc * 4);       // 16 MB
  float*    x1  = (float*)take((size_t)Bc * SQc * Dc * 4);       // 16 MB
  float*    x2  = (float*)take((size_t)Bc * SQc * Dc * 4);       // 16 MB
  float*    ffh = (float*)take((size_t)Bc * SQc * Fc * 4);       // 64 MB

  const int M  = Bc * SQc;   // 4096
  const int Mk = Bc * SKc;   // 8192
  const int BH = Bc * Hh;    // 128

  auto tpose = [&](const float* W, int K, int N) {
    transpose_cvt_kernel<<<dim3(K / 32, N / 32), 256, 0, stream>>>(W, wt, K, N);
  };
  auto gemm = [&](const float* A, const float* bias, float* C, int m, int n,
                  int k, int relu) {
    gemm_wmma_kernel<<<dim3(m / 64, n / 256), 256, 0, stream>>>(A, wt, bias, C,
                                                                m, n, k, relu);
  };

  // ---------------- Self-attention (causal) ----------------
  tpose(a1_wq, Dc, Dc); gemm(x, a1_bq, Qb, M, Dc, Dc, 0);
  tpose(a1_wk, Dc, Dc); gemm(x, a1_bk, Kb, M, Dc, Dc, 0);
  tpose(a1_wv, Dc, Dc); gemm(x, a1_bv, Vb, M, Dc, Dc, 0);
  vtranspose_kernel<<<dim3(SQc / 32, DKc / 32, BH), 256, 0, stream>>>(Vb, vt, SQc);
  attn_scores_kernel<<<dim3(SQc / 64, SQc / 128, BH), 256, 0, stream>>>(
      Qb, Kb, nullptr, Sb, SQc, SQc, /*causal=*/1);
  softmax_kernel<SQc><<<(BH * SQc) / 8, 256, 0, stream>>>(Sb);
  attn_av_kernel<<<dim3(SQc / 256, 1, BH), 256, 0, stream>>>(Sb, vt, ctx, SQc, SQc);
  tpose(a1_wo, Dc, Dc); gemm(ctx, a1_bo, res, M, Dc, Dc, 0);
  add_ln_kernel<<<M / 8, 256, 0, stream>>>(x, res, ln1_g, ln1_b, x1);

  // ---------------- Cross-attention (encoder mask) ----------------
  tpose(a2_wq, Dc, Dc); gemm(x1, a2_bq, Qb, M, Dc, Dc, 0);
  tpose(a2_wk, Dc, Dc); gemm(enc_o, a2_bk, Kb, Mk, Dc, Dc, 0);
  tpose(a2_wv, Dc, Dc); gemm(enc_o, a2_bv, Vb, Mk, Dc, Dc, 0);
  vtranspose_kernel<<<dim3(SKc / 32, DKc / 32, BH), 256, 0, stream>>>(Vb, vt, SKc);
  // Scores straight into d_out's attn region; softmax in place -> final probs.
  attn_scores_kernel<<<dim3(SQc / 64, SKc / 128, BH), 256, 0, stream>>>(
      Qb, Kb, encm, out_attn, SQc, SKc, /*causal=*/0);
  softmax_kernel<SKc><<<(BH * SQc) / 8, 256, 0, stream>>>(out_attn);
  attn_av_kernel<<<dim3(SQc / 256, 1, BH), 256, 0, stream>>>(out_attn, vt, ctx,
                                                             SQc, SKc);
  tpose(a2_wo, Dc, Dc); gemm(ctx, a2_bo, res, M, Dc, Dc, 0);
  add_ln_kernel<<<M / 8, 256, 0, stream>>>(x1, res, ln2_g, ln2_b, x2);

  // ---------------- FFN ----------------
  tpose(ff_w1, Dc, Fc); gemm(x2, ff_b1, ffh, M, Fc, Dc, /*relu=*/1);
  tpose(ff_w2, Fc, Dc); gemm(ffh, ff_b2, res, M, Dc, Fc, 0);
  add_ln_kernel<<<M / 8, 256, 0, stream>>>(x2, res, ln3_g, ln3_b, out_x);
}
